// EGNNLayer_10771777978566
// MI455X (gfx1250) — matrix operand order repaired and checked by reading, not compiled, not run on hardware
//
#include <hip/hip_runtime.h>
#include <hip/hip_bf16.h>
#include <stdint.h>

// ---------------- problem constants ----------------
#define HDIM   256
#define H2     512          // 2H
#define NNODE  50000
#define NEDGE  800000
#define EDIM   13
#define K1     526          // 2H + 1 + ED
#define K1P    544          // padded to 17*32 for WMMA K-loop
#define ET     32           // edges per wave tile (M=32 -> B-fragment reuse x2)

typedef __bf16 bf16;
typedef __attribute__((ext_vector_type(16))) __bf16 v16bf;
typedef __attribute__((ext_vector_type(8)))  float  v8f;

__device__ __forceinline__ float siluf(float x) { return x / (1.f + __expf(-x)); }
__device__ __forceinline__ float sigmf(float x) { return 1.f / (1.f + __expf(-x)); }

__device__ __forceinline__ void gatomic_add(float* p, float v) {
  unsafeAtomicAdd(p, v);   // global_atomic_add_f32
}

// CDNA5 async copy: 16B per lane, global -> LDS, tracked on ASYNCcnt.
// VDST = VGPR with LDS byte address (generic LDS addr truncates to offset),
// VADDR = 64-bit global address, GV mode (saddr = off).
__device__ __forceinline__ void async_copy16(uint32_t lds_addr, const void* gptr) {
  asm volatile("global_load_async_to_lds_b128 %0, %1, off"
               :: "v"(lds_addr), "v"(gptr) : "memory");
}
__device__ __forceinline__ void async_wait0() {
  asm volatile("s_wait_asynccnt 0x0" ::: "memory");
}

// A fragment: 16x32 bf16 tile from LDS, row-major, stride lda.
// lanes 0-15: M=lane, K={k0..k0+7, k0+16..k0+23}; lanes 16-31: M=lane-16, K={k0+8..k0+15, k0+24..k0+31}
__device__ __forceinline__ v16bf frag_a_lds(const bf16* base, int lda, int k0, int lane) {
  int m  = lane & 15;
  int kh = (lane >> 4) << 3;
  const bf16* p = base + m * lda + k0 + kh;
  v16bf a;
#pragma unroll
  for (int j = 0; j < 8; ++j) { a[j] = p[j]; a[j + 8] = p[j + 16]; }
  return a;
}

// B fragment: 32x16 bf16 from global weight stored transposed [N][Kpad] (K={kh..kh+7, kh+16..kh+23}, kh=8*(lane/16)).
__device__ __forceinline__ v16bf frag_b_glb(const bf16* Wt, int ldk, int n0, int k0, int lane) {
  int n  = lane & 15;
  int kh = (lane >> 4) << 3;
  const bf16* p = Wt + (size_t)(n0 + n) * ldk + k0 + kh;
  v16bf b;
#pragma unroll
  for (int j = 0; j < 8; ++j) { b[j] = p[j]; b[j + 8] = p[j + 16]; }
  return b;
}

__device__ __forceinline__ v8f wmma_bf16(v16bf a, v16bf b, v8f c) {
  return __builtin_amdgcn_wmma_f32_16x16x32_bf16(false, a, false, b, (short)0, c, false, false);
}

// ---------------- weight convert/transpose: f32 [K][Nf] -> bf16 [Nf][Kpad] ----------------
__global__ void k_transpose_bf16(const float* __restrict__ src, bf16* __restrict__ dst,
                                 int K, int Kpad, int Nf) {
  int idx = blockIdx.x * blockDim.x + threadIdx.x;
  if (idx >= Nf * Kpad) return;
  int n = idx / Kpad, k = idx % Kpad;
  float v = (k < K) ? src[(size_t)k * Nf + n] : 0.f;
  dst[idx] = (bf16)v;
}

__global__ void k_cvt_h(const float* __restrict__ h, bf16* __restrict__ hb, int n) {
  int i = blockIdx.x * blockDim.x + threadIdx.x;
  if (i < n) hb[i] = (bf16)h[i];
}

__global__ void k_zero_f32(float* __restrict__ p, int n) {
  int i = blockIdx.x * blockDim.x + threadIdx.x;
  if (i < n) p[i] = 0.f;
}

// ---------------- edge kernel: one wave32 per 32-edge tile (dynamic LDS ~68KB) ----------------
__global__ void __launch_bounds__(32) k_edge(
    const float* __restrict__ pos, const bf16* __restrict__ h_bf,
    const int* __restrict__ eidx, const float* __restrict__ eattr,
    const float* __restrict__ b_e1, const float* __restrict__ g_e,
    const float* __restrict__ be_ln, const float* __restrict__ b_e2,
    const float* __restrict__ W_a, const float* __restrict__ b_a,
    const float* __restrict__ b_c1, const float* __restrict__ W_c2,
    const bf16* __restrict__ We1t, const bf16* __restrict__ We2t,
    const bf16* __restrict__ Wc1t,
    float* __restrict__ agg, float* __restrict__ coord_agg) {
  extern __shared__ char smem[];
  bf16*  sb1   = (bf16*)smem;                        // [32][544] edge_in -> LN out [32][512]
  bf16*  sb2   = (bf16*)(smem + ET * K1P * 2);       // [32][512] act1 -> m [32][256]
  float* sdist = (float*)(smem + ET * K1P * 2 + ET * H2 * 2);
  float* sdiff = sdist + ET;                         // [32][3]
  float* satt  = sdiff + 3 * ET;
  float* scw   = satt + ET;
  int*   srow  = (int*)(scw + ET);
  int*   scol  = srow + ET;

  const int lane = threadIdx.x;
  const int e0 = blockIdx.x * ET;

  { // per-lane edge setup (all 32 lanes)
    int e = e0 + lane;
    int r = eidx[e], c = eidx[NEDGE + e];
    srow[lane] = r; scol[lane] = c;
    float dx = pos[r * 3 + 0] - pos[c * 3 + 0];
    float dy = pos[r * 3 + 1] - pos[c * 3 + 1];
    float dz = pos[r * 3 + 2] - pos[c * 3 + 2];
    sdiff[lane * 3 + 0] = dx; sdiff[lane * 3 + 1] = dy; sdiff[lane * 3 + 2] = dz;
    sdist[lane] = fmaxf(sqrtf(dx * dx + dy * dy + dz * dz), 1e-5f);
    satt[lane] = b_a[0];
    scw[lane] = 0.f;
  }
  __syncthreads();

  // edge_in rows: async bf16 copies of h[row]|h[col] (512B each), tail via VALU
  const uint32_t sb1_base = (uint32_t)(uintptr_t)sb1;
  for (int e = 0; e < ET; ++e) {
    const bf16* pr = h_bf + (size_t)srow[e] * HDIM;
    const bf16* pc = h_bf + (size_t)scol[e] * HDIM;
    uint32_t dst = sb1_base + (uint32_t)(e * K1P * 2) + lane * 16;
    async_copy16(dst, pr + lane * 8);
    async_copy16(dst + HDIM * 2, pc + lane * 8);
  }
  for (int e = 0; e < ET; ++e) {  // k = 512 + lane: [dist | 13 attrs | pad]
    float v = (lane == 0) ? sdist[e]
            : (lane < 1 + EDIM) ? eattr[(size_t)(e0 + e) * EDIM + (lane - 1)]
            : 0.f;
    sb1[e * K1P + H2 + lane] = (bf16)v;
  }
  async_wait0();
  __syncthreads();

  const int nl = lane & 15, khv = lane >> 4;

  // GEMM1: act1 = silu(edge_in @ W_e1 + b_e1) -> sb2 [32][512]; B reused for both M-halves
  for (int nt = 0; nt < H2 / 16; ++nt) {
    int n0 = nt * 16;
    float bias = b_e1[n0 + nl];
    v8f acc0, acc1;
#pragma unroll
    for (int v = 0; v < 8; ++v) { acc0[v] = bias; acc1[v] = bias; }
    for (int k0 = 0; k0 < K1P; k0 += 32) {
      v16bf b  = frag_b_glb(We1t, K1P, n0, k0, lane);
      v16bf a0 = frag_a_lds(sb1, K1P, k0, lane);
      v16bf a1 = frag_a_lds(sb1 + 16 * K1P, K1P, k0, lane);
      acc0 = wmma_bf16(a0, b, acc0);
      acc1 = wmma_bf16(a1, b, acc1);
    }
#pragma unroll
    for (int v = 0; v < 8; ++v) {
      int r = v + 8 * khv;
      sb2[r * H2 + n0 + nl]        = (bf16)siluf(acc0[v]);
      sb2[(r + 16) * H2 + n0 + nl] = (bf16)siluf(acc1[v]);
    }
  }
  __syncthreads();

  // LayerNorm(512) per edge row -> sb1 [32][512]
  for (int r = 0; r < ET; ++r) {
    float s = 0.f, s2 = 0.f;
    for (int k = lane; k < H2; k += 32) {
      float x = (float)sb2[r * H2 + k];
      s += x; s2 += x * x;
    }
#pragma unroll
    for (int off = 16; off > 0; off >>= 1) { s += __shfl_xor(s, off); s2 += __shfl_xor(s2, off); }
    float mu = s * (1.f / H2);
    float rstd = rsqrtf(s2 * (1.f / H2) - mu * mu + 1e-5f);
    for (int k = lane; k < H2; k += 32) {
      float x = (float)sb2[r * H2 + k];
      sb1[r * H2 + k] = (bf16)((x - mu) * rstd * g_e[k] + be_ln[k]);
    }
  }
  __syncthreads();

  // GEMM2: m = silu(ln @ W_e2 + b_e2) -> sb2 [32][256] bf16 ; fused att partial dot
  for (int nt = 0; nt < HDIM / 16; ++nt) {
    int n0 = nt * 16;
    float bias = b_e2[n0 + nl];
    v8f acc0, acc1;
#pragma unroll
    for (int v = 0; v < 8; ++v) { acc0[v] = bias; acc1[v] = bias; }
    for (int k0 = 0; k0 < H2; k0 += 32) {
      v16bf b  = frag_b_glb(We2t, H2, n0, k0, lane);
      v16bf a0 = frag_a_lds(sb1, H2, k0, lane);
      v16bf a1 = frag_a_lds(sb1 + 16 * H2, H2, k0, lane);
      acc0 = wmma_bf16(a0, b, acc0);
      acc1 = wmma_bf16(a1, b, acc1);
    }
    float wa = W_a[n0 + nl];
#pragma unroll
    for (int v = 0; v < 8; ++v) {
      int r = v + 8 * khv;
      float m0 = siluf(acc0[v]), m1 = siluf(acc1[v]);
      sb2[r * HDIM + n0 + nl]        = (bf16)m0;
      sb2[(r + 16) * HDIM + n0 + nl] = (bf16)m1;
      atomicAdd(&satt[r], m0 * wa);        // ds_add_f32
      atomicAdd(&satt[r + 16], m1 * wa);
    }
  }
  __syncthreads();
  satt[lane] = sigmf(satt[lane]);
  __syncthreads();

  // scatter agg += m * att
  for (int e = 0; e < ET; ++e) {
    float att = satt[e];
    float* dst = agg + (size_t)srow[e] * HDIM;
    for (int k = lane; k < HDIM; k += 32)
      gatomic_add(&dst[k], (float)sb2[e * HDIM + k] * att);
  }

  // coord MLP: cw = silu(m @ W_c1 + b_c1) @ W_c2 (fused dot)
  for (int nt = 0; nt < HDIM / 16; ++nt) {
    int n0 = nt * 16;
    float bias = b_c1[n0 + nl];
    v8f acc0, acc1;
#pragma unroll
    for (int v = 0; v < 8; ++v) { acc0[v] = bias; acc1[v] = bias; }
    for (int k0 = 0; k0 < HDIM; k0 += 32) {
      v16bf b  = frag_b_glb(Wc1t, HDIM, n0, k0, lane);
      v16bf a0 = frag_a_lds(sb2, HDIM, k0, lane);
      v16bf a1 = frag_a_lds(sb2 + 16 * HDIM, HDIM, k0, lane);
      acc0 = wmma_bf16(a0, b, acc0);
      acc1 = wmma_bf16(a1, b, acc1);
    }
    float wc = W_c2[n0 + nl];
#pragma unroll
    for (int v = 0; v < 8; ++v) {
      atomicAdd(&scw[v + 8 * khv], siluf(acc0[v]) * wc);
      atomicAdd(&scw[v + 8 * khv + 16], siluf(acc1[v]) * wc);
    }
  }
  __syncthreads();
  {
    float cw = scw[lane];
    float* dst = coord_agg + (size_t)srow[lane] * 3;
    gatomic_add(&dst[0], sdiff[lane * 3 + 0] * cw);
    gatomic_add(&dst[1], sdiff[lane * 3 + 1] * cw);
    gatomic_add(&dst[2], sdiff[lane * 3 + 2] * cw);
  }
}

#define EDGE_SMEM (ET * K1P * 2 + ET * H2 * 2 + (ET + 3 * ET + ET + ET) * 4 + 2 * ET * 4)

// ---------------- pos update ----------------
__global__ void k_pos(const float* __restrict__ pos, const float* __restrict__ coord_agg,
                      float* __restrict__ pout) {
  int i = blockIdx.x * blockDim.x + threadIdx.x;
  if (i < NNODE * 3) pout[i] = pos[i] + coord_agg[i];
}

// ---------------- node kernel: one wave32 per 16-node tile ----------------
__global__ void __launch_bounds__(32) k_node(
    const float* __restrict__ h, const bf16* __restrict__ h_bf,
    const float* __restrict__ agg,
    const float* __restrict__ b_n1, const float* __restrict__ g_n,
    const float* __restrict__ bn_ln, const float* __restrict__ b_n2,
    const float* __restrict__ g_o, const float* __restrict__ bo_ln,
    const bf16* __restrict__ Wn1t, const bf16* __restrict__ Wn2t,
    float* __restrict__ hout) {
  __shared__ bf16  sb1[16 * H2];
  __shared__ bf16  sb2[16 * H2];
  __shared__ float sh[16 * HDIM];
  const int lane = threadIdx.x;
  const int i0 = blockIdx.x * 16;

  // node_in = [h | agg]: h half via async bf16 row copies, agg half converted
  const uint32_t sb1_base = (uint32_t)(uintptr_t)sb1;
  for (int r = 0; r < 16; ++r)
    async_copy16(sb1_base + (uint32_t)(r * H2 * 2) + lane * 16,
                 h_bf + (size_t)(i0 + r) * HDIM + lane * 8);
  for (int r = 0; r < 16; ++r)
    for (int k = lane; k < HDIM; k += 32)
      sb1[r * H2 + HDIM + k] = (bf16)agg[(size_t)(i0 + r) * HDIM + k];
  async_wait0();
  __syncthreads();

  const int nl = lane & 15, khv = lane >> 4;

  // GEMM: silu(node_in @ W_n1 + b_n1) -> sb2
  for (int nt = 0; nt < H2 / 16; ++nt) {
    int n0 = nt * 16;
    float bias = b_n1[n0 + nl];
    v8f acc;
#pragma unroll
    for (int v = 0; v < 8; ++v) acc[v] = bias;
    for (int k0 = 0; k0 < H2; k0 += 32) {
      v16bf a = frag_a_lds(sb1, H2, k0, lane);
      v16bf b = frag_b_glb(Wn1t, H2, n0, k0, lane);
      acc = wmma_bf16(a, b, acc);
    }
#pragma unroll
    for (int v = 0; v < 8; ++v)
      sb2[(v + 8 * khv) * H2 + n0 + nl] = (bf16)siluf(acc[v]);
  }
  __syncthreads();

  // LayerNorm(512) -> sb1
  for (int r = 0; r < 16; ++r) {
    float s = 0.f, s2 = 0.f;
    for (int k = lane; k < H2; k += 32) {
      float x = (float)sb2[r * H2 + k];
      s += x; s2 += x * x;
    }
#pragma unroll
    for (int off = 16; off > 0; off >>= 1) { s += __shfl_xor(s, off); s2 += __shfl_xor(s2, off); }
    float mu = s * (1.f / H2);
    float rstd = rsqrtf(s2 * (1.f / H2) - mu * mu + 1e-5f);
    for (int k = lane; k < H2; k += 32) {
      float x = (float)sb2[r * H2 + k];
      sb1[r * H2 + k] = (bf16)((x - mu) * rstd * g_n[k] + bn_ln[k]);
    }
  }
  __syncthreads();

  // GEMM: upd = ln @ W_n2 + b_n2 ; residual h + upd -> sh (f32)
  for (int nt = 0; nt < HDIM / 16; ++nt) {
    int n0 = nt * 16;
    float bias = b_n2[n0 + nl];
    v8f acc;
#pragma unroll
    for (int v = 0; v < 8; ++v) acc[v] = bias;
    for (int k0 = 0; k0 < H2; k0 += 32) {
      v16bf a = frag_a_lds(sb1, H2, k0, lane);
      v16bf b = frag_b_glb(Wn2t, H2, n0, k0, lane);
      acc = wmma_bf16(a, b, acc);
    }
#pragma unroll
    for (int v = 0; v < 8; ++v) {
      int r = v + 8 * khv;
      sh[r * HDIM + n0 + nl] = h[(size_t)(i0 + r) * HDIM + n0 + nl] + acc[v];
    }
  }
  __syncthreads();

  // outer LayerNorm(256) -> hout
  for (int r = 0; r < 16; ++r) {
    float s = 0.f, s2 = 0.f;
    for (int k = lane; k < HDIM; k += 32) {
      float x = sh[r * HDIM + k];
      s += x; s2 += x * x;
    }
#pragma unroll
    for (int off = 16; off > 0; off >>= 1) { s += __shfl_xor(s, off); s2 += __shfl_xor(s2, off); }
    float mu = s * (1.f / HDIM);
    float rstd = rsqrtf(s2 * (1.f / HDIM) - mu * mu + 1e-5f);
    for (int k = lane; k < HDIM; k += 32)
      hout[(size_t)(i0 + r) * HDIM + k] = (sh[r * HDIM + k] - mu) * rstd * g_o[k] + bo_ln[k];
  }
}

// ---------------- feedback kernel: one wave32 per 16-edge tile ----------------
__global__ void __launch_bounds__(32) k_fb(
    const int* __restrict__ eidx, const float* __restrict__ pout,
    const float* __restrict__ W_f1, const float* __restrict__ b_f1,
    const float* __restrict__ b_f2, const bf16* __restrict__ Wf2t,
    float* __restrict__ hout) {
  __shared__ bf16  sv[16 * HDIM];
  __shared__ float sdist[16];
  __shared__ int   srow[16];
  const int lane = threadIdx.x;
  const int e0 = blockIdx.x * 16;

  if (lane < 16) {
    int e = e0 + lane;
    int r = eidx[e], c = eidx[NEDGE + e];
    srow[lane] = r;
    float dx = pout[r * 3 + 0] - pout[c * 3 + 0];
    float dy = pout[r * 3 + 1] - pout[c * 3 + 1];
    float dz = pout[r * 3 + 2] - pout[c * 3 + 2];
    sdist[lane] = fmaxf(sqrtf(dx * dx + dy * dy + dz * dz), 1e-5f);
  }
  __syncthreads();

  for (int e = 0; e < 16; ++e) {
    float d = sdist[e];
    for (int k = lane; k < HDIM; k += 32)
      sv[e * HDIM + k] = (bf16)siluf(d * W_f1[k] + b_f1[k]);
  }
  __syncthreads();

  const int nl = lane & 15, khv = lane >> 4;
  for (int nt = 0; nt < HDIM / 16; ++nt) {
    int n0 = nt * 16;
    float bias = b_f2[n0 + nl];
    v8f acc;
#pragma unroll
    for (int v = 0; v < 8; ++v) acc[v] = bias;
    for (int k0 = 0; k0 < HDIM; k0 += 32) {
      v16bf a = frag_a_lds(sv, HDIM, k0, lane);
      v16bf b = frag_b_glb(Wf2t, HDIM, n0, k0, lane);
      acc = wmma_bf16(a, b, acc);
    }
#pragma unroll
    for (int v = 0; v < 8; ++v) {
      int r = v + 8 * khv;
      gatomic_add(&hout[(size_t)srow[r] * HDIM + n0 + nl], 0.1f * acc[v]);
    }
  }
}

// ---------------- host launch ----------------
extern "C" void kernel_launch(void* const* d_in, const int* in_sizes, int n_in,
                              void* d_out, int out_size, void* d_ws, size_t ws_size,
                              hipStream_t stream) {
  const float* h     = (const float*)d_in[0];
  const float* pos   = (const float*)d_in[1];
  const int*   eidx  = (const int*)d_in[2];
  const float* eattr = (const float*)d_in[3];
  const float* W_e1  = (const float*)d_in[4];
  const float* b_e1  = (const float*)d_in[5];
  const float* g_e   = (const float*)d_in[6];
  const float* be_ln = (const float*)d_in[7];
  const float* W_e2  = (const float*)d_in[8];
  const float* b_e2  = (const float*)d_in[9];
  const float* W_n1  = (const float*)d_in[10];
  const float* b_n1  = (const float*)d_in[11];
  const float* g_n   = (const float*)d_in[12];
  const float* bn_ln = (const float*)d_in[13];
  const float* W_n2  = (const float*)d_in[14];
  const float* b_n2  = (const float*)d_in[15];
  const float* g_o   = (const float*)d_in[16];
  const float* bo_ln = (const float*)d_in[17];
  const float* W_c1  = (const float*)d_in[18];
  const float* b_c1  = (const float*)d_in[19];
  const float* W_c2  = (const float*)d_in[20];
  const float* W_a   = (const float*)d_in[21];
  const float* b_a   = (const float*)d_in[22];
  const float* W_f1  = (const float*)d_in[23];
  const float* b_f1  = (const float*)d_in[24];
  const float* W_f2  = (const float*)d_in[25];
  const float* b_f2  = (const float*)d_in[26];

  char* ws = (char*)d_ws;
  size_t off = 0;
  bf16* We1t = (bf16*)(ws + off); off += (size_t)H2 * K1P * 2;        // [512][544]
  bf16* We2t = (bf16*)(ws + off); off += (size_t)HDIM * H2 * 2;       // [256][512]
  bf16* Wn1t = (bf16*)(ws + off); off += (size_t)H2 * H2 * 2;         // [512][512]
  bf16* Wn2t = (bf16*)(ws + off); off += (size_t)HDIM * H2 * 2;       // [256][512]
  bf16* Wc1t = (bf16*)(ws + off); off += (size_t)HDIM * HDIM * 2;     // [256][256]
  bf16* Wf2t = (bf16*)(ws + off); off += (size_t)HDIM * HDIM * 2;     // [256][256]
  off = (off + 255) & ~(size_t)255;
  bf16* h_bf = (bf16*)(ws + off); off += (size_t)NNODE * HDIM * 2;    // bf16 copy of h
  off = (off + 255) & ~(size_t)255;
  float* agg = (float*)(ws + off); off += (size_t)NNODE * HDIM * 4;
  float* coord_agg = (float*)(ws + off); off += (size_t)NNODE * 3 * 4;

  float* hout = (float*)d_out;                 // [N,256]
  float* pout = hout + (size_t)NNODE * HDIM;   // [N,3]

  auto gr = [](int n) { return (n + 255) / 256; };

  // one-time conversions (weights are L2-resident afterwards)
  k_transpose_bf16<<<gr(H2 * K1P),    256, 0, stream>>>(W_e1, We1t, K1,  K1P, H2);
  k_transpose_bf16<<<gr(HDIM * H2),   256, 0, stream>>>(W_e2, We2t, H2,  H2,  HDIM);
  k_transpose_bf16<<<gr(H2 * H2),     256, 0, stream>>>(W_n1, Wn1t, H2,  H2,  H2);
  k_transpose_bf16<<<gr(HDIM * H2),   256, 0, stream>>>(W_n2, Wn2t, H2,  H2,  HDIM);
  k_transpose_bf16<<<gr(HDIM * HDIM), 256, 0, stream>>>(W_c1, Wc1t, HDIM, HDIM, HDIM);
  k_transpose_bf16<<<gr(HDIM * HDIM), 256, 0, stream>>>(W_f2, Wf2t, HDIM, HDIM, HDIM);
  k_cvt_h<<<gr(NNODE * HDIM), 256, 0, stream>>>(h, h_bf, NNODE * HDIM);

  int nzero = NNODE * HDIM + NNODE * 3;        // agg + coord_agg contiguous
  k_zero_f32<<<gr(nzero), 256, 0, stream>>>(agg, nzero);

  // per-edge fused MLPs + scatters (32 edges / wave, dynamic LDS)
  k_edge<<<NEDGE / ET, 32, EDGE_SMEM, stream>>>(pos, h_bf, eidx, eattr,
                                                b_e1, g_e, be_ln, b_e2, W_a, b_a, b_c1, W_c2,
                                                We1t, We2t, Wc1t, agg, coord_agg);

  k_pos<<<gr(NNODE * 3), 256, 0, stream>>>(pos, coord_agg, pout);

  k_node<<<NNODE / 16, 32, 0, stream>>>(h, h_bf, agg, b_n1, g_n, bn_ln, b_n2, g_o, bo_ln,
                                        Wn1t, Wn2t, hout);

  k_fb<<<NEDGE / 16, 32, 0, stream>>>(eidx, pout, W_f1, b_f1, b_f2, Wf2t, hout);

  (void)in_sizes; (void)n_in; (void)out_size; (void)ws_size;
}
